// Decoder_46574625357933
// MI455X (gfx1250) — matrix operand order repaired and checked by reading, not compiled, not run on hardware
//
#include <hip/hip_runtime.h>
#include <math.h>

typedef __attribute__((ext_vector_type(16))) _Float16 v16h;
typedef __attribute__((ext_vector_type(8)))  _Float16 half8;
typedef __attribute__((ext_vector_type(4)))  _Float16 half4;
typedef __attribute__((ext_vector_type(8)))  float    v8f;
typedef __attribute__((ext_vector_type(4)))  unsigned int uint4v;
typedef __attribute__((ext_vector_type(8)))  unsigned int uint8v;

#define H_EDGES 100000
#define S_MEM   16
#define D_DIM   128
#define HID_DIM 128
#define W_STRIDE 136            // padded LDS row stride (halves): 128 data + 8 pad
#define WAVES   4
#define TPB     (WAVES * 32)
#define EDGES_PER_WAVE 16
#define EDGES_PER_BLOCK (WAVES * EDGES_PER_WAVE)

// ---------------------------------------------------------------------------
// Pre-kernel: W1 [128x128] f32 row-major  ->  W1^T [n][k] f16 dense in d_ws.
// Runs once per launch; 64 KB read, 32 KB written. Amortized over 1563 blocks.
// ---------------------------------------------------------------------------
__global__ __launch_bounds__(256)
void w1_transpose_f16_kernel(const float* __restrict__ W1, _Float16* __restrict__ w1t)
{
    const int idx = blockIdx.x * 256 + threadIdx.x;     // 16384 threads
    if (idx < D_DIM * HID_DIM) {
        const int k = idx >> 7;      // row of W1 (D)
        const int n = idx & 127;     // col of W1 (HID)
        w1t[n * D_DIM + k] = (_Float16)W1[idx];         // coalesced read
    }
}

// ---------------------------------------------------------------------------
// Main kernel: TDM-stages W1^T(f16) into padded LDS (overlapped with gather),
// max-min pools gathered rows into f16 A tiles, then WMMA GEMM + fused MLP head.
// ---------------------------------------------------------------------------
__global__ __launch_bounds__(TPB)
void hyperedge_maxmin_mlp_kernel(const float* __restrict__ v_feat,
                                 const int*   __restrict__ hedge_info,
                                 const _Float16* __restrict__ w1t,   // pre-transposed f16
                                 const float* __restrict__ b1,
                                 const float* __restrict__ W2,
                                 const float* __restrict__ b2,
                                 float*       __restrict__ out)
{
    __shared__ _Float16 lds_w1t[HID_DIM * W_STRIDE];   // filled by TDM with HW padding
    __shared__ _Float16 lds_a[WAVES * 16 * W_STRIDE];  // per-wave 16x128 A tiles (f16)

    const int tid      = threadIdx.x;
    const int lane     = tid & 31;
    const int wid      = tid >> 5;
    const int half_sel = lane >> 4;   // 0 = lanes 0-15, 1 = lanes 16-31
    const int l16      = lane & 15;

    // ---- Stage 0: wave 0 issues one TDM load of W1^T (32 KB) into LDS ----
    // D# (CDNA5 ISA): 2D tile 128x128 of 2-byte elements,
    // pad_enable: after every 64 DWORDs (one 128-half row) insert 4 DWORDs
    // (8 halves) of padding -> LDS stride 136 halves, matching W_STRIDE.
    if (wid == 0) {
        const unsigned long long ga = (unsigned long long)(size_t)w1t;
        const unsigned ldsoff = (unsigned)(size_t)(void*)lds_w1t;  // low 32 bits = LDS byte offset

        uint4v g0;
        g0[0] = 1u;                                   // count=1, user descriptor
        g0[1] = ldsoff;                               // lds_addr [63:32]
        g0[2] = (unsigned)(ga & 0xFFFFFFFFu);         // global_addr low
        g0[3] = (unsigned)((ga >> 32) & 0x1FFFFFFu)   // global_addr high (57-bit)
              | 0x80000000u;                          // type=2 ("image") in bits[127:126]

        uint8v g1;
        g1[0] = (1u << 16)      // data_size = 1 (2 bytes)
              | (1u << 20)      // pad_enable
              | (5u << 22)      // pad_interval = 5 -> 64 DWORDs (one 256B row)
              | (3u << 25);     // pad_amount   = 3 -> 4 DWORDs (16B = 8 halves)
        g1[1] = (128u << 16);   // tensor_dim0 = 128 (bits 79:48, low half)
        g1[2] = (128u << 16);   // tensor_dim0 hi = 0 | tensor_dim1 = 128 (low half)
        g1[3] = (128u << 16);   // tensor_dim1 hi = 0 | tile_dim0 = 128
        g1[4] = 128u;           // tile_dim1 = 128, tile_dim2 = 0
        g1[5] = 128u;           // tensor_dim0_stride = 128 (elements)
        g1[6] = 0u;
        g1[7] = 0u;

        asm volatile("tensor_load_to_lds %0, %1" :: "s"(g0), "s"(g1) : "memory");
    }

    // ---- Stage 1: gather + max-min pool -> A tile (f16) in LDS (overlaps TDM) ----
    const int mBase = blockIdx.x * EDGES_PER_BLOCK + wid * EDGES_PER_WAVE;
    _Float16* aTile = lds_a + wid * 16 * W_STRIDE;

    for (int e = 0; e < EDGES_PER_WAVE; ++e) {
        int hg = mBase + e;
        if (hg >= H_EDGES) hg = H_EDGES - 1;          // clamp: loads stay valid, EXEC stays full
        const int* he = hedge_info + (size_t)hg * S_MEM;

        float4 mx, mn;
        {
            const float4 v = *(const float4*)(v_feat + (size_t)he[0] * D_DIM + lane * 4);
            mx = v; mn = v;
        }
        #pragma unroll
        for (int s = 1; s < S_MEM; ++s) {
            const float4 v = *(const float4*)(v_feat + (size_t)he[s] * D_DIM + lane * 4);
            mx.x = fmaxf(mx.x, v.x); mn.x = fminf(mn.x, v.x);
            mx.y = fmaxf(mx.y, v.y); mn.y = fminf(mn.y, v.y);
            mx.z = fmaxf(mx.z, v.z); mn.z = fminf(mn.z, v.z);
            mx.w = fmaxf(mx.w, v.w); mn.w = fminf(mn.w, v.w);
        }
        half4 agg;
        agg[0] = (_Float16)(mx.x - mn.x);
        agg[1] = (_Float16)(mx.y - mn.y);
        agg[2] = (_Float16)(mx.z - mn.z);
        agg[3] = (_Float16)(mx.w - mn.w);
        *(half4*)(aTile + e * W_STRIDE + lane * 4) = agg;   // ds_store_b64
    }

    // wave 0 waits for the TDM transfer; barrier publishes LDS to all waves
    if (wid == 0) __builtin_amdgcn_s_wait_tensorcnt(0);
    __syncthreads();

    // ---- Stage 2: load A fragments (ISA 16-bit A layout, K in 4 chunks of 32) ----
    // lanes 0-15 (row = l16): K = k0 + {0..7, 16..23};  lanes 16-31: K = k0 + {8..15, 24..31}
    v16h afrag[4];
    {
        const _Float16* arow = aTile + l16 * W_STRIDE;
        #pragma unroll
        for (int kt = 0; kt < 4; ++kt) {
            const int k0 = kt * 32 + half_sel * 8;
            const half8 c0 = *(const half8*)(arow + k0);        // ds_load_b128
            const half8 c1 = *(const half8*)(arow + k0 + 16);   // ds_load_b128
            #pragma unroll
            for (int i = 0; i < 8; ++i) { afrag[kt][i] = c0[i]; afrag[kt][8 + i] = c1[i]; }
        }
    }

    // ---- Stage 3: h = relu(A @ W1 + b1) fused with preds = sigmoid(h @ W2 + b2) ----
    float part[8];
    #pragma unroll
    for (int i = 0; i < 8; ++i) part[i] = 0.f;

    #pragma unroll
    for (int nt = 0; nt < 8; ++nt) {
        const int n = nt * 16 + l16;
        // B fragment: lane's column n; lanes 0-15: K = k0+{0..7,16..23}, lanes 16-31: K = k0+{8..15,24..31}
        const _Float16* brow = lds_w1t + n * W_STRIDE + half_sel * 8;

        v8f acc = {};
        #pragma unroll
        for (int kt = 0; kt < 4; ++kt) {
            const half8 c0 = *(const half8*)(brow + kt * 32);
            const half8 c1 = *(const half8*)(brow + kt * 32 + 16);
            v16h bfrag;
            #pragma unroll
            for (int i = 0; i < 8; ++i) { bfrag[i] = c0[i]; bfrag[8 + i] = c1[i]; }

            acc = __builtin_amdgcn_wmma_f32_16x16x32_f16(
                /*neg_a=*/false, afrag[kt], /*neg_b=*/false, bfrag,
                /*c_mod=*/(short)0, acc, /*reuse_a=*/false, /*reuse_b=*/false);
        }

        // epilogue for this 16-column tile: bias + ReLU + scale by W2[n], accumulate
        const float b1n = b1[n];
        const float w2n = W2[n];
        #pragma unroll
        for (int i = 0; i < 8; ++i) {
            float h = acc[i] + b1n;          // C layout: VGPR i, lanes<16 -> M=i, lanes>=16 -> M=i+8
            h = fmaxf(h, 0.f);
            part[i] += h * w2n;
        }
    }

    // ---- Stage 4: per-row reduction across the 16 lanes holding each row's columns ----
    #pragma unroll
    for (int off = 8; off >= 1; off >>= 1) {
        #pragma unroll
        for (int i = 0; i < 8; ++i)
            part[i] += __shfl_xor(part[i], off, 32);   // stays within each 16-lane half
    }

    const float bias2 = b2[0];
    if (l16 == 0) {                                    // lane 0 -> rows 0..7, lane 16 -> rows 8..15
        #pragma unroll
        for (int i = 0; i < 8; ++i) {
            const int m = mBase + half_sel * 8 + i;
            if (m < H_EDGES)
                out[m] = 1.0f / (1.0f + expf(-(part[i] + bias2)));
        }
    }
}

extern "C" void kernel_launch(void* const* d_in, const int* in_sizes, int n_in,
                              void* d_out, int out_size, void* d_ws, size_t ws_size,
                              hipStream_t stream) {
    (void)in_sizes; (void)n_in; (void)ws_size; (void)out_size;
    const float* v_feat     = (const float*)d_in[0];
    const int*   hedge_info = (const int*)  d_in[1];
    const float* W1         = (const float*)d_in[2];
    const float* b1         = (const float*)d_in[3];
    const float* W2         = (const float*)d_in[4];
    const float* b2         = (const float*)d_in[5];
    float*       out        = (float*)d_out;

    _Float16* w1t = (_Float16*)d_ws;   // 32 KB scratch: W1^T in f16

    w1_transpose_f16_kernel<<<(D_DIM * HID_DIM + 255) / 256, 256, 0, stream>>>(W1, w1t);

    const int grid = (H_EDGES + EDGES_PER_BLOCK - 1) / EDGES_PER_BLOCK;  // 1563
    hyperedge_maxmin_mlp_kernel<<<grid, TPB, 0, stream>>>(
        v_feat, hedge_info, w1t, b1, W2, b2, out);
}
